// Head_48722109005851
// MI455X (gfx1250) — matrix operand order repaired and checked by reading, not compiled, not run on hardware
//
#include <hip/hip_runtime.h>
#include <stdint.h>

#define BATCH 4
#define SEQ   4096
#define EMBD  1024
#define HS    64
#define BT    (BATCH * SEQ)          // 16384 rows

typedef __attribute__((ext_vector_type(16))) __bf16 v16bf;
typedef __attribute__((ext_vector_type(8)))  float  v8f;

// fp32 -> bf16 (round-to-nearest-even), raw bits
static __device__ __forceinline__ unsigned short f2bf(float f) {
  union { float f; unsigned u; } x; x.f = f;
  unsigned u = x.u;
  u += 0x7FFFu + ((u >> 16) & 1u);
  return (unsigned short)(u >> 16);
}
// load a 16-element bf16 fragment from a 16B-aligned address (2x ds_load_b128)
static __device__ __forceinline__ v16bf ld_frag(const unsigned short* p) {
  union { v16bf v; uint4 u[2]; } t;
  t.u[0] = *reinterpret_cast<const uint4*>(p);
  t.u[1] = *reinterpret_cast<const uint4*>(p + 8);
  return t.v;
}
static __device__ __forceinline__ v16bf ld_frag2(const unsigned short* p0,
                                                 const unsigned short* p1) {
  union { v16bf v; uint4 u[2]; } t;
  t.u[0] = *reinterpret_cast<const uint4*>(p0);
  t.u[1] = *reinterpret_cast<const uint4*>(p1);
  return t.v;
}
// CDNA5 async DMA: global -> LDS, 16 bytes, tracked by ASYNCcnt
static __device__ __forceinline__ void async_b128(const unsigned short* lds,
                                                  const unsigned short* g) {
  unsigned loff = (unsigned)(uintptr_t)lds;
  unsigned long long ga = (unsigned long long)(uintptr_t)g;
  asm volatile("global_load_async_to_lds_b128 %0, %1, off"
               :: "v"(loff), "v"(ga) : "memory");
}
// async loads complete in order: waiting <=N forces the oldest batch done
#define WAIT_ASYNC(n) asm volatile("s_wait_asynccnt " #n ::: "memory")

// ---------------- prep: x fp32 -> bf16 (row-major) -----------------------------
__global__ __launch_bounds__(256) void cvt_x(const float* __restrict__ x,
                                             unsigned short* __restrict__ xb) {
  size_t i = ((size_t)blockIdx.x * 256 + threadIdx.x) * 8;
  float4 a = *reinterpret_cast<const float4*>(x + i);
  float4 b = *reinterpret_cast<const float4*>(x + i + 4);
  unsigned short o[8] = { f2bf(a.x), f2bf(a.y), f2bf(a.z), f2bf(a.w),
                          f2bf(b.x), f2bf(b.y), f2bf(b.z), f2bf(b.w) };
  *reinterpret_cast<uint4*>(xb + i) = *reinterpret_cast<const uint4*>(o);
}

// ---------------- prep: W [C][HS] fp32 -> WT [HS][C] bf16 ----------------------
__global__ __launch_bounds__(256) void cvt_w_t(const float* __restrict__ Wq,
                                               const float* __restrict__ Wk,
                                               const float* __restrict__ Wv,
                                               unsigned short* __restrict__ WTq,
                                               unsigned short* __restrict__ WTk,
                                               unsigned short* __restrict__ WTv) {
  int gid = blockIdx.x * 256 + threadIdx.x;       // 3 * 1024 * 64 total
  int mat = gid >> 16;
  int rem = gid & 65535;
  int k = rem >> 6, n = rem & 63;
  const float* W = (mat == 0) ? Wq : (mat == 1) ? Wk : Wv;
  unsigned short* WT = (mat == 0) ? WTq : (mat == 1) ? WTk : WTv;
  WT[(size_t)n * EMBD + k] = f2bf(W[(size_t)k * HS + n]);
}

// ---------------- QKV projection (bf16 WMMA), q pre-scaled, V transposed -------
// Double-buffered LDS, source-level unroll-by-2 so each accumulator set has a
// stable register home; async DMA for step kk+1 overlaps WMMAs on step kk.
__global__ __launch_bounds__(256) void qkv_proj_wmma(
    const unsigned short* __restrict__ xb,
    const unsigned short* __restrict__ WTq, const float* __restrict__ bq,
    const unsigned short* __restrict__ WTk, const float* __restrict__ bk,
    const unsigned short* __restrict__ WTv, const float* __restrict__ bv,
    unsigned short* __restrict__ qo, unsigned short* __restrict__ ko,
    unsigned short* __restrict__ vTo)
{
  const unsigned short* WT; const float* bias; unsigned short* out;
  if (blockIdx.y == 0)      { WT = WTq; bias = bq; out = qo;  }
  else if (blockIdx.y == 1) { WT = WTk; bias = bk; out = ko;  }
  else                      { WT = WTv; bias = bv; out = vTo; }
  const bool isQ = (blockIdx.y == 0);
  const bool isV = (blockIdx.y == 2);

  // x tile staged in A-fragment order: [strip][rlane][e] -> contiguous per lane
  __shared__ alignas(32) unsigned short xs[2 * 8 * 512];
  // W tile transposed: [n][k] stride 40 (16B-aligned fragment rows)
  __shared__ alignas(32) unsigned short wt[2 * 64 * 40];

  const int tid  = threadIdx.x;
  const int wave = tid >> 5;
  const int lane = tid & 31;
  const int half = lane >> 4;
  const int l15  = lane & 15;
  const int rowbase = blockIdx.x * 128;

  // stage step kk into buffer buf: 3 async b128 per thread
  auto stage = [&](int kk, int buf) {
    #pragma unroll
    for (int i = 0; i < 2; ++i) {
      int ci = tid + i * 256;
      int row = ci >> 2, j = ci & 3;            // j: 8-element k-chunk
      const unsigned short* src = xb + (size_t)(rowbase + row) * EMBD + kk * 32 + j * 8;
      const unsigned short* dst =
          &xs[buf * 4096 + (row >> 4) * 512 + ((row & 15) + 16 * (j & 1)) * 16 + 8 * (j >> 1)];
      async_b128(dst, src);
    }
    int n = tid >> 2, jc = tid & 3;
    async_b128(&wt[buf * 2560 + n * 40 + jc * 8], WT + (size_t)n * EMBD + kk * 32 + jc * 8);
  };

  v8f acc[4] = {};
  auto mma_step = [&](int buf) {
    v16bf a = ld_frag(&xs[buf * 4096 + wave * 512 + lane * 16]);
    #pragma unroll
    for (int nt = 0; nt < 4; ++nt) {
      v16bf bm = ld_frag2(&wt[buf * 2560 + (nt * 16 + l15) * 40 + 8 * half],
                          &wt[buf * 2560 + (nt * 16 + l15) * 40 + 8 * half + 16]);
      acc[nt] = __builtin_amdgcn_wmma_f32_16x16x32_bf16(
          false, a, false, bm, (short)0, acc[nt], false, false);
    }
  };

  stage(0, 0);
  for (int kk = 0; kk < EMBD / 32; kk += 2) {
    // even half: compute buf0, fetch kk+1 into buf1
    stage(kk + 1, 1);
    WAIT_ASYNC(3);
    __syncthreads();
    mma_step(0);
    __syncthreads();
    // odd half: compute buf1, fetch kk+2 into buf0
    if (kk + 2 < EMBD / 32) { stage(kk + 2, 0); WAIT_ASYNC(3); }
    else                    { WAIT_ASYNC(0); }
    __syncthreads();
    mma_step(1);
    __syncthreads();
  }

  const float sc = isQ ? 0.03125f : 1.0f;       // fold C^-0.5 into q
  #pragma unroll
  for (int nt = 0; nt < 4; ++nt) {
    int n = nt * 16 + l15;
    float bb = bias[n];
    #pragma unroll
    for (int r = 0; r < 8; ++r) {
      int row = rowbase + wave * 16 + r + 8 * half;   // C/D layout: m = r + 8*half
      unsigned short val = f2bf((acc[nt][r] + bb) * sc);
      if (isV) out[(size_t)n * BT + row] = val;       // V stored transposed [h][BT]
      else     out[(size_t)row * HS + n] = val;
    }
  }
}

// ---------------- Flash attention: 128 q-rows / block, 64-key tiles ------------
// Double-buffered K/V tiles; nkt is always even, so the loop unrolls by 2
// cleanly; async DMA for tile t+1 overlaps tile t compute.
__global__ __launch_bounds__(256) void flash_attn_wmma(
    const unsigned short* __restrict__ qg,
    const unsigned short* __restrict__ kg,
    const unsigned short* __restrict__ vTg,
    float* __restrict__ outg)
{
  __shared__ alignas(32) unsigned short kt[2 * 64 * 72];  // [key][h]
  __shared__ alignas(32) unsigned short vt[2 * 64 * 72];  // [h][key] (transposed)
  __shared__ alignas(32) unsigned short pt[8 * 512];      // one 32-K P chunk/wave

  const int tid  = threadIdx.x;
  const int wave = tid >> 5;
  const int lane = tid & 31;
  const int half = lane >> 4;
  const int l15  = lane & 15;

  const int b  = blockIdx.y;
  const int qb = blockIdx.x * 128;
  const int qrow0 = qb + wave * 16;

  const unsigned short* qm = qg + (size_t)b * SEQ * HS;
  const unsigned short* km = kg + (size_t)b * SEQ * HS;
  float* om = outg + (size_t)b * SEQ * HS;

  // stage 64-key tile into buffer buf: 4 async b128 per thread
  auto stage = [&](int kb, int buf) {
    #pragma unroll
    for (int i = 0; i < 2; ++i) {
      int idx = tid + i * 256;
      int r = idx >> 3, c8 = (idx & 7) * 8;
      async_b128(&kt[buf * 4608 + r * 72 + c8], km + (size_t)(kb + r) * HS + c8);
      async_b128(&vt[buf * 4608 + r * 72 + c8], vTg + (size_t)r * BT + b * SEQ + kb + c8);
    }
  };

  // q fragments straight from global: two contiguous 16B chunks each
  v16bf qf[2];
  {
    const unsigned short* qrow = qm + (size_t)(qrow0 + l15) * HS;
    #pragma unroll
    for (int c = 0; c < 2; ++c)
      qf[c] = ld_frag2(qrow + c * 32 + 8 * half, qrow + c * 32 + 8 * half + 16);
  }

  float mrow[8], lrow[8];
  #pragma unroll
  for (int r = 0; r < 8; ++r) { mrow[r] = -3.0e38f; lrow[r] = 0.f; }
  v8f oacc[4] = {};

  // writer-side P slot math (constant per lane)
  const int p_e_lo = 2 * ((l15 >> 1) & 3) + (l15 & 1);    // + 8*(nt&1)
  const int p_rl16 = 16 * ((l15 >> 3) & 1);               // rlane bias

  // process one staged 64-key tile
  auto tile_step = [&](int kb, int buf) {
    if (kb > qrow0 + 15) return;     // wave-uniform: strip entirely above tile
    const unsigned short* ktb = &kt[buf * 4608];
    const unsigned short* vtb = &vt[buf * 4608];

    // S = q . K^T
    v8f s[4] = {};
    #pragma unroll
    for (int nt = 0; nt < 4; ++nt)
      #pragma unroll
      for (int c = 0; c < 2; ++c) {
        v16bf kf = ld_frag2(&ktb[(nt * 16 + l15) * 72 + c * 32 + 8 * half],
                            &ktb[(nt * 16 + l15) * 72 + c * 32 + 8 * half + 16]);
        s[nt] = __builtin_amdgcn_wmma_f32_16x16x32_bf16(
            false, qf[c], false, kf, (short)0, s[nt], false, false);
      }

    // causal mask only on diagonal tiles
    if (kb + 63 > qrow0) {
      #pragma unroll
      for (int nt = 0; nt < 4; ++nt)
        #pragma unroll
        for (int r = 0; r < 8; ++r) {
          int row = qrow0 + r + 8 * half;
          int col = kb + nt * 16 + l15;
          if (col > row) s[nt][r] = -3.0e38f;
        }
    }

    // online softmax (rows live in 16-lane half-groups)
    float rmax[8];
    #pragma unroll
    for (int r = 0; r < 8; ++r)
      rmax[r] = fmaxf(fmaxf(s[0][r], s[1][r]), fmaxf(s[2][r], s[3][r]));
    #pragma unroll
    for (int r = 0; r < 8; ++r)
      #pragma unroll
      for (int msk = 1; msk < 16; msk <<= 1)
        rmax[r] = fmaxf(rmax[r], __shfl_xor(rmax[r], msk, 32));

    float corr[8], radd[8];
    #pragma unroll
    for (int r = 0; r < 8; ++r) {
      float mn = fmaxf(mrow[r], rmax[r]);
      corr[r] = exp2f((mrow[r] - mn) * 1.4426950408889634f);
      mrow[r] = mn;
      radd[r] = 0.f;
    }
    #pragma unroll
    for (int nt = 0; nt < 4; ++nt)
      #pragma unroll
      for (int r = 0; r < 8; ++r) {
        float p = exp2f((s[nt][r] - mrow[r]) * 1.4426950408889634f);
        s[nt][r] = p;
        radd[r] += p;
      }
    #pragma unroll
    for (int r = 0; r < 8; ++r) {
      #pragma unroll
      for (int msk = 1; msk < 16; msk <<= 1)
        radd[r] += __shfl_xor(radd[r], msk, 32);
      lrow[r] = lrow[r] * corr[r] + radd[r];
    }
    #pragma unroll
    for (int nt = 0; nt < 4; ++nt)
      #pragma unroll
      for (int r = 0; r < 8; ++r)
        oacc[nt][r] *= corr[r];

    // O += P @ V, one 32-K chunk at a time (wave-private staging, DScnt-ordered)
    #pragma unroll
    for (int c = 0; c < 2; ++c) {
      #pragma unroll
      for (int ntc = 0; ntc < 2; ++ntc) {       // the two 16-col tiles in chunk c
        int nt = c * 2 + ntc;
        int base = wave * 512 + 8 * ntc + p_e_lo;
        #pragma unroll
        for (int r = 0; r < 8; ++r)
          pt[base + (r + 8 * half + p_rl16) * 16] = f2bf(s[nt][r]);
      }
      v16bf pf = ld_frag(&pt[wave * 512 + lane * 16]);
      #pragma unroll
      for (int nt = 0; nt < 4; ++nt) {
        v16bf vf = ld_frag2(&vtb[(nt * 16 + l15) * 72 + c * 32 + 8 * half],
                            &vtb[(nt * 16 + l15) * 72 + c * 32 + 8 * half + 16]);
        oacc[nt] = __builtin_amdgcn_wmma_f32_16x16x32_bf16(
            false, pf, false, vf, (short)0, oacc[nt], false, false);
      }
    }
  };

  const int nkt = (qb + 128) >> 6;   // = 2*(blockIdx.x+1), always even
  stage(0, 0);
  for (int t = 0; t < nkt; t += 2) {
    // even tile: compute buf0, fetch t+1 into buf1 (t+1 < nkt always holds)
    stage((t + 1) * 64, 1);
    WAIT_ASYNC(4);
    __syncthreads();
    tile_step(t * 64, 0);
    __syncthreads();
    // odd tile: compute buf1, fetch t+2 into buf0
    if (t + 2 < nkt) { stage((t + 2) * 64, 0); WAIT_ASYNC(4); }
    else             { WAIT_ASYNC(0); }
    __syncthreads();
    tile_step((t + 1) * 64, 1);
    __syncthreads();
  }

  #pragma unroll
  for (int nt = 0; nt < 4; ++nt) {
    int n = nt * 16 + l15;
    #pragma unroll
    for (int r = 0; r < 8; ++r) {
      int row = qrow0 + r + 8 * half;
      om[(size_t)row * HS + n] = oacc[nt][r] / lrow[r];
    }
  }
}

extern "C" void kernel_launch(void* const* d_in, const int* in_sizes, int n_in,
                              void* d_out, int out_size, void* d_ws, size_t ws_size,
                              hipStream_t stream) {
  (void)in_sizes; (void)n_in; (void)out_size; (void)ws_size;
  const float* x  = (const float*)d_in[0];
  const float* Wq = (const float*)d_in[1];
  const float* bq = (const float*)d_in[2];
  const float* Wk = (const float*)d_in[3];
  const float* bk = (const float*)d_in[4];
  const float* Wv = (const float*)d_in[5];
  const float* bv = (const float*)d_in[6];
  float* out = (float*)d_out;

  const size_t N = (size_t)BT * HS;            // 1 Mi elements per q/k/v
  unsigned short* qw  = (unsigned short*)d_ws;
  unsigned short* kw  = qw + N;
  unsigned short* vTw = kw + N;
  unsigned short* xb  = vTw + N;               // BT*EMBD bf16 = 33.5 MB
  unsigned short* WTq = xb + (size_t)BT * EMBD;
  unsigned short* WTk = WTq + (size_t)EMBD * HS;
  unsigned short* WTv = WTk + (size_t)EMBD * HS;

  cvt_x<<<dim3((unsigned)((size_t)BT * EMBD / (256 * 8))), 256, 0, stream>>>(x, xb);
  cvt_w_t<<<dim3(3 * EMBD * HS / 256), 256, 0, stream>>>(Wq, Wk, Wv, WTq, WTk, WTv);
  qkv_proj_wmma<<<dim3(BT / 128, 3), 256, 0, stream>>>(
      xb, WTq, bq, WTk, bk, WTv, bv, qw, kw, vTw);
  flash_attn_wmma<<<dim3(SEQ / 128, BATCH), 256, 0, stream>>>(qw, kw, vTw, out);
}
